// LTFDyG_41738492182951
// MI455X (gfx1250) — matrix operand order repaired and checked by reading, not compiled, not run on hardware
//
#include <hip/hip_runtime.h>
#include <hip/hip_bf16.h>

#define NUM_IDS 10001
#define B_SZ 128
#define L_SZ 1024
#define D_SZ 128
#define T_SZ 100
#define ROWS_PER_SIDE (B_SZ * L_SZ) /* 131072 rows per side */
#define M_BLOCK 128                 /* rows per workgroup   */

typedef __attribute__((ext_vector_type(16))) _Float16 v16h;
typedef __attribute__((ext_vector_type(8)))  _Float16 v8h;
typedef __attribute__((ext_vector_type(8)))  float    v8f;

// ---------------------------------------------------------------------------
// K1: zero the histogram workspace (harness poisons d_ws; must re-zero per call)
// ---------------------------------------------------------------------------
__global__ void ltf_zero_ws(float* __restrict__ p, int n) {
  int i = blockIdx.x * blockDim.x + threadIdx.x;
  if (i < n) p[i] = 0.0f;
}

// ---------------------------------------------------------------------------
// K2: per-batch histograms via float atomics (adds of exactly 1.0f -> the
// result is order-independent and exact; deterministic across replays).
// ---------------------------------------------------------------------------
__global__ void ltf_hist(const int* __restrict__ src_ids,
                         const int* __restrict__ dst_ids,
                         float* __restrict__ hist_src,
                         float* __restrict__ hist_dst) {
  int i = blockIdx.x * blockDim.x + threadIdx.x;
  if (i >= ROWS_PER_SIDE) return;
  int b = i >> 10;                       // L = 1024
  int s = src_ids[i];
  int d = dst_ids[i];
  if (s != 0) atomicAdd(&hist_src[b * NUM_IDS + s], 1.0f);
  if (d != 0) atomicAdd(&hist_dst[b * NUM_IDS + d], 1.0f);
}

// ---------------------------------------------------------------------------
// K3: fused gather + time-weight + MLP-as-GEMM (WMMA f16 -> f32 accum).
// grid = (ROWS_PER_SIDE / M_BLOCK, 2), block = 256 (8 wave32's).
// blockIdx.y: 0 = src side, 1 = dst side.
// out[b,l,:] = (relu(x0*W1+b1) + relu(x1*W1+b1)) @ W2 + 2*b2
// ---------------------------------------------------------------------------
__launch_bounds__(256)
__global__ void ltf_main(const int*   __restrict__ src_ids,
                         const int*   __restrict__ dst_ids,
                         const float* __restrict__ src_times,
                         const float* __restrict__ dst_times,
                         const float* __restrict__ nit,
                         const float* __restrict__ w_time,
                         const float* __restrict__ b_time,
                         const float* __restrict__ W_ts,
                         const float* __restrict__ b_ts,
                         const float* __restrict__ W1,
                         const float* __restrict__ b1,
                         const float* __restrict__ W2,
                         const float* __restrict__ b2,
                         const float* __restrict__ hist_src,
                         const float* __restrict__ hist_dst,
                         float* __restrict__ out) {
  // 32 KB + 32 KB = 64 KB LDS
  __shared__ _Float16 w2t[D_SZ * D_SZ];    // [n][k] : W2 transposed, f16
  __shared__ _Float16 gsh[M_BLOCK * D_SZ]; // [row][k]: activation tile, f16

  const int tid  = threadIdx.x;
  const int side = blockIdx.y;
  const int row0 = blockIdx.x * M_BLOCK;

  const int*   ids = side ? dst_ids : src_ids;
  const float* tms = side ? dst_times : src_times;
  float* outp = out + (size_t)side * ROWS_PER_SIDE * D_SZ;

  // ---- Phase 0: stage W2^T into LDS as f16 -------------------------------
  for (int e = tid; e < D_SZ * D_SZ; e += 256) {
    int n = e >> 7;          // e = n*128 + k
    int k = e & (D_SZ - 1);
    w2t[e] = (_Float16)W2[k * D_SZ + n];
  }

  // ---- Phase 1: build the activation tile g (f16) ------------------------
  {
    const int rlocal = tid & (M_BLOCK - 1); // row within block
    const int khalf  = tid >> 7;            // each row split across 2 threads
    const int grow   = row0 + rlocal;       // == b*L + l
    const int b      = grow >> 10;

    const int   id = ids[grow];
    const float dt = nit[b] - tms[grow];

    // time weight: sigmoid( cos(dt*w_time + b_time) . W_ts + b_ts )
    float s = b_ts[0];
    #pragma unroll 4
    for (int j = 0; j < T_SZ; ++j)
      s += __cosf(dt * w_time[j] + b_time[j]) * W_ts[j];
    const float w = 1.0f / (1.0f + __expf(-s));

    // gathers (bin 0 is always 0.0, matching the reference's padding rule)
    const float x0 = hist_src[b * NUM_IDS + id] * w;
    const float x1 = hist_dst[b * NUM_IDS + id] * w;

    const int k0 = khalf * (D_SZ / 2);
    #pragma unroll 8
    for (int k = k0; k < k0 + D_SZ / 2; ++k) {
      const float w1k = W1[k];
      const float b1k = b1[k];
      const float g = fmaxf(__builtin_fmaf(x0, w1k, b1k), 0.0f) +
                      fmaxf(__builtin_fmaf(x1, w1k, b1k), 0.0f);
      gsh[rlocal * D_SZ + k] = (_Float16)g;
    }
  }
  __syncthreads();

  // ---- Phase 2: (128 x 128) @ (128 x 128) GEMM via WMMA ------------------
  const int wave = tid >> 5;
  const int lane = tid & 31;
  const int t16  = lane & 15; // A-row / B-col / D-col within 16x16 tile
  const int hsel = lane >> 4; // lane half selects K-subgroup / M-subgroup

  v8f acc[8] = {}; // 8 N-tiles of 16, f32 accumulators

  const int arow = wave * 16 + t16;
  #pragma unroll
  for (int kt = 0; kt < 4; ++kt) {
    // A fragment (contiguous k order, same as B): lane half 0 -> K {kt*32+0..15}
    //                                   lane half 1 -> K {kt*32+16..31}
    const int kbase = kt * 32 + hsel * 16;
    union { v16h v; v8h h[2]; } a;
    a.h[0] = *(const v8h*)&gsh[arow * D_SZ + kbase];
    a.h[1] = *(const v8h*)&gsh[arow * D_SZ + kbase + 8];

    #pragma unroll
    for (int nt = 0; nt < 8; ++nt) {
      // B fragment (32x16): lane half h holds K = kt*32 + 16*h .. +15, col t16;
      // contiguous 32B in w2t[n][k] layout.
      const v16h bf =
          *(const v16h*)&w2t[(nt * 16 + t16) * D_SZ + kt * 32 + hsel * 16];
      acc[nt] = __builtin_amdgcn_wmma_f32_16x16x32_f16(
          /*neg_a=*/false, a.v, /*neg_b=*/false, bf,
          /*c_mod=*/(short)0, acc[nt], /*reuse_a=*/false, /*reuse_b=*/false);
    }
  }

  // ---- Epilogue: + 2*b2, store (C/D layout: VGPR r -> M = r + 8*hsel) ----
  #pragma unroll
  for (int nt = 0; nt < 8; ++nt) {
    const int n = nt * 16 + t16;
    const float bias2 = 2.0f * b2[n];
    #pragma unroll
    for (int r = 0; r < 8; ++r) {
      const int grow = row0 + wave * 16 + r + 8 * hsel;
      outp[(size_t)grow * D_SZ + n] = acc[nt][r] + bias2;
    }
  }
}

// ---------------------------------------------------------------------------
extern "C" void kernel_launch(void* const* d_in, const int* in_sizes, int n_in,
                              void* d_out, int out_size, void* d_ws, size_t ws_size,
                              hipStream_t stream) {
  const int*   src_ids   = (const int*)  d_in[0];
  const int*   dst_ids   = (const int*)  d_in[1];
  const float* src_times = (const float*)d_in[2];
  const float* dst_times = (const float*)d_in[3];
  const float* nit       = (const float*)d_in[4];
  const float* w_time    = (const float*)d_in[5];
  const float* b_time    = (const float*)d_in[6];
  const float* W_ts      = (const float*)d_in[7];
  const float* b_ts      = (const float*)d_in[8];
  const float* W1        = (const float*)d_in[9];
  const float* b1        = (const float*)d_in[10];
  const float* W2        = (const float*)d_in[11];
  const float* b2        = (const float*)d_in[12];
  float* out = (float*)d_out;

  float* hist_src = (float*)d_ws;
  float* hist_dst = hist_src + (size_t)B_SZ * NUM_IDS;

  const int histN = 2 * B_SZ * NUM_IDS; // ~9.8 MB of f32 in workspace
  ltf_zero_ws<<<(histN + 255) / 256, 256, 0, stream>>>((float*)d_ws, histN);

  ltf_hist<<<(ROWS_PER_SIDE + 255) / 256, 256, 0, stream>>>(
      src_ids, dst_ids, hist_src, hist_dst);

  dim3 grid(ROWS_PER_SIDE / M_BLOCK, 2);
  ltf_main<<<grid, 256, 0, stream>>>(src_ids, dst_ids, src_times, dst_times,
                                     nit, w_time, b_time, W_ts, b_ts,
                                     W1, b1, W2, b2, hist_src, hist_dst, out);
}
